// InteractionPPBlockSMP_32384053412123
// MI455X (gfx1250) — matrix operand order repaired and checked by reading, not compiled, not run on hardware
//
#include <hip/hip_runtime.h>

// DimeNet++ InteractionPPBlock for MI455X (gfx1250, wave32, WMMA bf16).
// Weights pre-packed into WMMA B-fragment layout (1x b256 load / fragment),
// activations carried in bf16 row-major (2x b128 loads / A fragment).

#define E_EDGES 65536
#define T_TRIP  262144

typedef __attribute__((ext_vector_type(16))) __bf16 v16bf;
typedef __attribute__((ext_vector_type(8)))  __bf16 v8bf;
typedef __attribute__((ext_vector_type(8)))  float  v8f;

__device__ __forceinline__ float silu_f(float v) {
  return v * (1.0f / (1.0f + __expf(-v)));
}

// A-fragment (16x32 bf16) from row-major bf16 (matches packed B):
// lane<16: K=k0+0..15 ; lane>=16: K=k0+16..31.
// Two aligned 16B vector loads.
__device__ __forceinline__ v16bf load_frag_a_bf(const __bf16* __restrict__ A, int lda,
                                                int row0, int k0, int lane) {
  const int l = lane & 15, hi = lane >> 4;
  const __bf16* p = A + (size_t)(row0 + l) * lda + k0 + hi * 16;
  const v8bf lo = *(const v8bf*)p;
  const v8bf hv = *(const v8bf*)(p + 8);
  return __builtin_shufflevector(lo, hv, 0, 1, 2, 3, 4, 5, 6, 7,
                                 8, 9, 10, 11, 12, 13, 14, 15);
}

// B-fragment from pre-packed weights: fragment (kt,nt) stores lane L's 16
// bf16 values contiguously -> one aligned 32B load per lane.
__device__ __forceinline__ v16bf load_frag_b_packed(const __bf16* __restrict__ Wp,
                                                    int ntn, int kt, int nt, int lane) {
  return *(const v16bf*)(Wp + ((size_t)(kt * ntn + nt) * 32 + lane) * 16);
}

// Pack W[K,N] (f32 row-major) into WMMA B-fragment bf16 layout.
// lane<16 holds K=kt*32+0..15, lane>=16 K=kt*32+16..31, col = nt*16+(lane&15).
__global__ void __launch_bounds__(256)
pack_w_kernel(const float* __restrict__ W, int K, int N, __bf16* __restrict__ out) {
  const int idx = blockIdx.x * 256 + threadIdx.x;
  const int total = (K >> 5) * (N >> 4) * 32;
  if (idx >= total) return;
  const int lane = idx & 31;
  const int tile = idx >> 5;
  const int ntn = N >> 4;
  const int nt = tile % ntn;
  const int kt = tile / ntn;
  const int l = lane & 15, hi = lane >> 4;
  const int col = nt * 16 + l;
  const int kb = kt * 32 + hi * 16;
  v16bf f;
#pragma unroll
  for (int e = 0; e < 16; ++e) f[e] = (__bf16)W[(size_t)(kb + e) * N + col];
  *(v16bf*)(out + (size_t)idx * 16) = f;
}

__global__ void __launch_bounds__(256)
f32_to_bf16_kernel(const float* __restrict__ in, __bf16* __restrict__ out, size_t n) {
  const size_t i = (size_t)blockIdx.x * 256 + threadIdx.x;
  if (i < n) out[i] = (__bf16)in[i];
}

// Generic fused GEMM: v = silu(A@W + bias) [* rbf_p]; out = (res?res:0)+v
// written as f32 (out_f32) and/or bf16 (out_bf). N in {64,128}.
// Block = 256 threads = 8 waves; wave tile = 16 rows x 64 cols (4 acc).
template <int N, bool RBF>
__global__ void __launch_bounds__(256)
gemm_silu_kernel(const __bf16* __restrict__ A, int K,
                 const __bf16* __restrict__ Wp,    // packed fragments
                 const float* __restrict__ bias,   // [N] or null
                 const float* __restrict__ res,    // [rows,N] f32 or null
                 const float* __restrict__ rbf,    // [E,6]   (RBF only)
                 const float* __restrict__ wrbf,   // [6,N]   (RBF only)
                 float* __restrict__ out_f32,      // [rows,N] or null
                 __bf16* __restrict__ out_bf) {    // [rows,N] or null
  constexpr int COLG   = N / 64;
  constexpr int BROWS  = (8 / COLG) * 16;   // 64 (N=128) or 128 (N=64)
  constexpr int LDSROW = N + 2;
  constexpr int NTN    = N / 16;
  __shared__ float srbf[RBF ? BROWS * LDSROW : 1];

  const int lane    = threadIdx.x & 31;
  const int w       = threadIdx.x >> 5;
  const int rowTile = (w / COLG) * 16;
  const int colTile = (w % COLG) * 64;
  const int rowBase = blockIdx.x * BROWS;

  if (RBF) {
    for (int idx = threadIdx.x; idx < BROWS * N; idx += 256) {
      const int r = idx / N, c = idx - r * N;
      const float* rp = rbf + (size_t)(rowBase + r) * 6;
      float v = 0.f;
#pragma unroll
      for (int q = 0; q < 6; ++q) v += rp[q] * wrbf[q * N + c];
      srbf[r * LDSROW + c] = v;
    }
    __syncthreads();
  }

  v8f acc[4];
#pragma unroll
  for (int n = 0; n < 4; ++n)
#pragma unroll
    for (int i = 0; i < 8; ++i) acc[n][i] = 0.f;

  for (int k0 = 0, kt = 0; k0 < K; k0 += 32, ++kt) {
    const v16bf a = load_frag_a_bf(A, K, rowBase + rowTile, k0, lane);
#pragma unroll
    for (int n = 0; n < 4; ++n) {
      const v16bf b = load_frag_b_packed(Wp, NTN, kt, (colTile >> 4) + n, lane);
      acc[n] = __builtin_amdgcn_wmma_f32_16x16x32_bf16(
          false, a, false, b, (short)0, acc[n], false, false);
    }
  }

  const int l = lane & 15, hi = lane >> 4;
#pragma unroll
  for (int n = 0; n < 4; ++n) {
    const int col = colTile + n * 16 + l;
#pragma unroll
    for (int r = 0; r < 8; ++r) {
      const int rib = rowTile + hi * 8 + r;
      const size_t row = (size_t)rowBase + rib;
      float v = acc[n][r];
      if (bias) v += bias[col];
      v = silu_f(v);
      if (RBF) v *= srbf[rib * LDSROW + col];
      if (res) v += res[row * N + col];
      if (out_f32) out_f32[row * N + col] = v;
      if (out_bf) out_bf[row * N + col] = (__bf16)v;
    }
  }
}

// Collapse rbf chain: Wrbf_c[b,r,g] = sum_c W_rbf1[b,r,c]*W_rbf2[b,c,g]
__global__ void __launch_bounds__(256)
collapse_wrbf_kernel(const float* __restrict__ w1, const float* __restrict__ w2,
                     float* __restrict__ out) {
  const int idx = blockIdx.x * 256 + threadIdx.x;
  if (idx >= 6 * 6 * 128) return;
  const int g = idx & 127;
  const int r = (idx >> 7) % 6;
  const int b = idx / (6 * 128);
  float v = 0.f;
#pragma unroll
  for (int c = 0; c < 8; ++c)
    v += w1[(b * 6 + r) * 8 + c] * w2[(b * 8 + c) * 128 + g];
  out[idx] = v;
}

__global__ void __launch_bounds__(256)
zero_kernel(float* __restrict__ p, size_t n) {
  const size_t i = (size_t)blockIdx.x * 256 + threadIdx.x;
  if (i < n) p[i] = 0.f;
}

// Per-triplet fused sbf-projection + gather + weighted scatter-add.
// Only branch 5 (weight alpha) and branch b=bt[idx_kj]+1 (weight 1-alpha)
// contribute per triplet (BT_LIST structure).
__global__ void __launch_bounds__(256)
scatter_kernel(const float* __restrict__ sbf,     // [T,42]
               const int* __restrict__ idx_kj,
               const int* __restrict__ idx_ji,
               const int* __restrict__ bt,
               const float* __restrict__ Wsbf1,   // [6,42,8]
               const float* __restrict__ Wsbf2,   // [6,8,64]
               const __bf16* __restrict__ down,   // [6,E,64] bf16
               const float* __restrict__ alpha_p, // [1]
               float* __restrict__ xkj) {         // [E,64] accumulators
  __shared__ float sW1[6 * 42 * 8];
  __shared__ float sW2[6 * 8 * 64];
  for (int i = threadIdx.x; i < 6 * 42 * 8; i += 256) sW1[i] = Wsbf1[i];
  for (int i = threadIdx.x; i < 6 * 8 * 64; i += 256) sW2[i] = Wsbf2[i];
  __syncthreads();

  const int t = blockIdx.x * 256 + threadIdx.x;
  if (t >= T_TRIP) return;

  const float alpha = alpha_p[0];
  const int j = idx_kj[t];
  const int i = idx_ji[t];
  const int bv = bt[j];
  int b = bv + 1;                               // BT_LIST[b] == bv
  const bool match = (b >= 0) && (b <= 5);
  if (!match) b = 0;
  const float bw = match ? (1.0f - alpha) : 0.0f;

  float s[42];
  const float* sp = sbf + (size_t)t * 42;
#pragma unroll
  for (int k = 0; k < 42; ++k) s[k] = sp[k];

  float c5[8], cb[8];
#pragma unroll
  for (int c = 0; c < 8; ++c) { c5[c] = 0.f; cb[c] = 0.f; }
  const float* w5 = sW1 + 5 * 42 * 8;
  const float* wb = sW1 + b * 42 * 8;
  for (int k = 0; k < 42; ++k) {
    const float sv = s[k];
#pragma unroll
    for (int c = 0; c < 8; ++c) {
      c5[c] += sv * w5[k * 8 + c];
      cb[c] += sv * wb[k * 8 + c];
    }
  }

  const float* m5 = sW2 + 5 * 8 * 64;
  const float* mb = sW2 + b * 8 * 64;
  const __bf16* d5 = down + ((size_t)5 * E_EDGES + j) * 64;
  const __bf16* db = down + ((size_t)b * E_EDGES + j) * 64;
  float* dst = xkj + (size_t)i * 64;
  for (int d0 = 0; d0 < 64; d0 += 8) {
    const v8bf g5 = *(const v8bf*)(d5 + d0);
    const v8bf gb = *(const v8bf*)(db + d0);
#pragma unroll
    for (int e = 0; e < 8; ++e) {
      const int d = d0 + e;
      float s5 = 0.f, sb = 0.f;
#pragma unroll
      for (int c = 0; c < 8; ++c) {
        s5 += c5[c] * m5[c * 64 + d];
        sb += cb[c] * mb[c * 64 + d];
      }
      const float v = alpha * ((float)g5[e] * s5) + bw * ((float)gb[e] * sb);
      atomicAdd(dst + d, v);
    }
  }
}

extern "C" void kernel_launch(void* const* d_in, const int* in_sizes, int n_in,
                              void* d_out, int out_size, void* d_ws, size_t ws_size,
                              hipStream_t stream) {
  (void)in_sizes; (void)n_in; (void)out_size; (void)ws_size;
  const float* x      = (const float*)d_in[0];
  const float* rbf    = (const float*)d_in[1];
  const float* sbf    = (const float*)d_in[2];
  const int*   idx_kj = (const int*)d_in[3];
  const int*   idx_ji = (const int*)d_in[4];
  const int*   bt     = (const int*)d_in[5];
  /* d_in[6] = lambda_d (unused by reference) */
  const float* alpha  = (const float*)d_in[7];
  const float* W_kj   = (const float*)d_in[8];
  const float* b_kj   = (const float*)d_in[9];
  const float* W_rbf1 = (const float*)d_in[10];
  const float* W_rbf2 = (const float*)d_in[11];
  const float* W_sbf1 = (const float*)d_in[12];
  const float* W_sbf2 = (const float*)d_in[13];
  const float* W_down = (const float*)d_in[14];
  const float* W_ji   = (const float*)d_in[15];
  const float* b_ji   = (const float*)d_in[16];
  const float* W_up   = (const float*)d_in[17];
  const float* rb1_w  = (const float*)d_in[18];
  const float* rb1_b  = (const float*)d_in[19];
  const float* rb2_w  = (const float*)d_in[20];
  const float* rb2_b  = (const float*)d_in[21];
  const float* W_lin  = (const float*)d_in[22];
  const float* b_lin  = (const float*)d_in[23];
  const float* ra1_w  = (const float*)d_in[24];
  const float* ra1_b  = (const float*)d_in[25];
  const float* ra2_w  = (const float*)d_in[26];
  const float* ra2_b  = (const float*)d_in[27];

  // Workspace carve-out (256B aligned chunks), total ~210 MB.
  char* wp = (char*)d_ws;
  auto alloc = [&](size_t bytes) -> char* {
    char* r = wp;
    wp += (bytes + 255) & ~(size_t)255;
    return r;
  };
  const size_t EH = (size_t)E_EDGES * 128;
  float*  xji     = (float*)alloc(EH * 4);
  float*  hbuf    = (float*)alloc(EH * 4);
  float*  xkj     = (float*)alloc((size_t)E_EDGES * 64 * 4);
  float*  wrbfc   = (float*)alloc(6 * 6 * 128 * 4);
  __bf16* x_bf    = (__bf16*)alloc(EH * 2);
  __bf16* tmp_bf  = (__bf16*)alloc(EH * 2);
  __bf16* hbuf_bf = (__bf16*)alloc(EH * 2);
  __bf16* h2_bf   = (__bf16*)alloc(EH * 2);
  __bf16* xkj_bf  = (__bf16*)alloc((size_t)E_EDGES * 64 * 2);
  __bf16* down_bf = (__bf16*)alloc((size_t)6 * E_EDGES * 64 * 2);
  __bf16* Wji_p   = (__bf16*)alloc(128 * 128 * 2);
  __bf16* Wkj_p   = (__bf16*)alloc((size_t)6 * 128 * 128 * 2);
  __bf16* Wdown_p = (__bf16*)alloc((size_t)6 * 128 * 64 * 2);
  __bf16* Wup_p   = (__bf16*)alloc(64 * 128 * 2);
  __bf16* rb1_p   = (__bf16*)alloc(128 * 128 * 2);
  __bf16* rb2_p   = (__bf16*)alloc(128 * 128 * 2);
  __bf16* lin_p   = (__bf16*)alloc(128 * 128 * 2);
  __bf16* ra1_p   = (__bf16*)alloc(128 * 128 * 2);
  __bf16* ra2_p   = (__bf16*)alloc(128 * 128 * 2);

  const dim3 blk(256);
  const dim3 g128(E_EDGES / 64);
  const dim3 g64(E_EDGES / 128);
  const dim3 gp128x128((128 / 32) * (128 / 16) * 32 / 256);  // 4 blocks
  const dim3 gp128x64(((128 / 32) * (64 / 16) * 32 + 255) / 256);
  const dim3 gp64x128(((64 / 32) * (128 / 16) * 32 + 255) / 256);

  // ---- weight prep ----
  collapse_wrbf_kernel<<<dim3((6 * 6 * 128 + 255) / 256), blk, 0, stream>>>(
      W_rbf1, W_rbf2, wrbfc);
  pack_w_kernel<<<gp128x128, blk, 0, stream>>>(W_ji, 128, 128, Wji_p);
  for (int b = 0; b < 6; ++b) {
    pack_w_kernel<<<gp128x128, blk, 0, stream>>>(
        W_kj + (size_t)b * 128 * 128, 128, 128, Wkj_p + (size_t)b * 128 * 128);
    pack_w_kernel<<<gp128x64, blk, 0, stream>>>(
        W_down + (size_t)b * 128 * 64, 128, 64, Wdown_p + (size_t)b * 128 * 64);
  }
  pack_w_kernel<<<gp64x128, blk, 0, stream>>>(W_up, 64, 128, Wup_p);
  pack_w_kernel<<<gp128x128, blk, 0, stream>>>(rb1_w, 128, 128, rb1_p);
  pack_w_kernel<<<gp128x128, blk, 0, stream>>>(rb2_w, 128, 128, rb2_p);
  pack_w_kernel<<<gp128x128, blk, 0, stream>>>(W_lin, 128, 128, lin_p);
  pack_w_kernel<<<gp128x128, blk, 0, stream>>>(ra1_w, 128, 128, ra1_p);
  pack_w_kernel<<<gp128x128, blk, 0, stream>>>(ra2_w, 128, 128, ra2_p);
  f32_to_bf16_kernel<<<dim3(EH / 256), blk, 0, stream>>>(x, x_bf, EH);

  // ---- x_ji = silu(x @ W_ji + b_ji) ----
  gemm_silu_kernel<128, false><<<g128, blk, 0, stream>>>(
      x_bf, 128, Wji_p, b_ji, nullptr, nullptr, nullptr, xji, nullptr);

  // ---- per branch: tmp = silu(x@W_kj+b_kj)*rbf_p ; down = silu(tmp@W_down) ----
  for (int b = 0; b < 6; ++b) {
    gemm_silu_kernel<128, true><<<g128, blk, 0, stream>>>(
        x_bf, 128, Wkj_p + (size_t)b * 128 * 128, b_kj + b * 128, nullptr,
        rbf, wrbfc + b * 6 * 128, nullptr, tmp_bf);
    gemm_silu_kernel<64, false><<<g64, blk, 0, stream>>>(
        tmp_bf, 128, Wdown_p + (size_t)b * 128 * 64, nullptr, nullptr,
        nullptr, nullptr, nullptr, down_bf + (size_t)b * E_EDGES * 64);
  }

  // ---- triplet scatter into x_kj_tot ----
  zero_kernel<<<dim3((E_EDGES * 64) / 256), blk, 0, stream>>>(
      xkj, (size_t)E_EDGES * 64);
  scatter_kernel<<<dim3(T_TRIP / 256), blk, 0, stream>>>(
      sbf, idx_kj, idx_ji, bt, W_sbf1, W_sbf2, down_bf, alpha, xkj);
  f32_to_bf16_kernel<<<dim3((E_EDGES * 64) / 256), blk, 0, stream>>>(
      xkj, xkj_bf, (size_t)E_EDGES * 64);

  // ---- h = x_ji + silu(x_kj_tot @ W_up) ----
  gemm_silu_kernel<128, false><<<g128, blk, 0, stream>>>(
      xkj_bf, 64, Wup_p, nullptr, xji, nullptr, nullptr, hbuf, hbuf_bf);

  // ---- before-skip residual block ----
  gemm_silu_kernel<128, false><<<g128, blk, 0, stream>>>(
      hbuf_bf, 128, rb1_p, rb1_b, nullptr, nullptr, nullptr, nullptr, tmp_bf);
  gemm_silu_kernel<128, false><<<g128, blk, 0, stream>>>(
      tmp_bf, 128, rb2_p, rb2_b, hbuf, nullptr, nullptr, nullptr, h2_bf);

  // ---- h = silu(h @ W_lin + b_lin) + x ----
  gemm_silu_kernel<128, false><<<g128, blk, 0, stream>>>(
      h2_bf, 128, lin_p, b_lin, x, nullptr, nullptr, hbuf, hbuf_bf);

  // ---- after-skip residual block -> d_out ----
  gemm_silu_kernel<128, false><<<g128, blk, 0, stream>>>(
      hbuf_bf, 128, ra1_p, ra1_b, nullptr, nullptr, nullptr, nullptr, tmp_bf);
  gemm_silu_kernel<128, false><<<g128, blk, 0, stream>>>(
      tmp_bf, 128, ra2_p, ra2_b, hbuf, nullptr, nullptr, (float*)d_out, nullptr);
}
